// FGNN_46531675685504
// MI455X (gfx1250) — matrix operand order repaired and checked by reading, not compiled, not run on hardware
//
#include <hip/hip_runtime.h>
#include <hip/hip_bf16.h>

#define NPIX  65536
#define CIN   32
#define COUT  32
#define NRING 23

typedef __attribute__((ext_vector_type(16))) _Float16 v16h;
typedef __attribute__((ext_vector_type(8)))  float    v8f;

// A-matrix (16x32 f16) K slot for element i, half-select hs = lane>>4.
// ISA 7.12.2 16-bit A table: VGPR v holds K = {0..7,16..23} + 8*hs, 2 per VGPR.
__device__ __forceinline__ int kmapA(int i, int hs) {
  int v = i >> 1;
  return ((v & 4) << 2) + ((v & 3) << 1) + (hs << 3) + (i & 1);
}
// B-matrix (32x16 f16) K slot: same order as A:
// K = {0..7,16..23} + 8*hs, 2 per VGPR.
__device__ __forceinline__ int kmapB(int i, int hs) { return kmapA(i, hs); }

__device__ __forceinline__ v8f wmma16(v16h a, v16h b, v8f c) {
  return __builtin_amdgcn_wmma_f32_16x16x32_f16(false, a, false, b, (short)0, c,
                                                false, false);
}

// split-f16 product: (Ah+Al)(Bh+Bl) ~= AhBh + AhBl + AlBh (near-fp32 accuracy)
__device__ __forceinline__ v8f wmma3(v16h ah, v16h al, v16h bh, v16h bl, v8f c) {
  c = wmma16(ah, bh, c);
  c = wmma16(ah, bl, c);
  c = wmma16(al, bh, c);
  return c;
}

template <bool BM, typename F>
__device__ __forceinline__ void frag_split(F f, int lane, v16h& hi, v16h& lo) {
  int a = lane & 15, hs = (lane >> 4) & 1;
#pragma unroll
  for (int i = 0; i < 16; ++i) {
    float x = f(a, BM ? kmapB(i, hs) : kmapA(i, hs));
    _Float16 h = (_Float16)x;
    hi[i] = h;
    lo[i] = (_Float16)(x - (float)h);
  }
}

template <bool BM, typename F>
__device__ __forceinline__ v16h frag16(F f, int lane) {
  int a = lane & 15, hs = (lane >> 4) & 1;
  v16h o;
#pragma unroll
  for (int i = 0; i < 16; ++i)
    o[i] = (_Float16)f(a, BM ? kmapB(i, hs) : kmapA(i, hs));
  return o;
}

// ---------------- small utility kernels ----------------
__global__ void k_zero(float* p, int n) {
  int i = blockIdx.x * blockDim.x + threadIdx.x;
  int st = gridDim.x * blockDim.x;
  for (; i < n; i += st) p[i] = 0.0f;
}

__global__ void k_count(const int* __restrict__ ring_id, float* cnt) {
  int i = blockIdx.x * blockDim.x + threadIdx.x;
  int st = gridDim.x * blockDim.x;
  for (; i < NPIX; i += st) atomicAdd(&cnt[ring_id[i]], 1.0f);
}

__global__ void k_mean(float* s_re, float* s_im, const float* cnt) {
  int i = blockIdx.x * blockDim.x + threadIdx.x;
  if (i >= NRING * COUT) return;
  float c = cnt[i >> 5];
  c = (c > 0.0f) ? c : 1.0f;
  s_re[i] /= c;
  s_im[i] /= c;
}

__global__ void k_sigmoid(float* g, int n) {
  int i = blockIdx.x * blockDim.x + threadIdx.x;
  int st = gridDim.x * blockDim.x;
  for (; i < n; i += st) g[i] = 1.0f / (1.0f + __expf(-g[i]));
}

// ---------------- stage 1: u1 = (x*unc)@W1, c = (x*cor)@W2 ----------------
// one wave per 16-pixel tile; ring sums of u1 via fp32 atomics;
// hb = unc ? 0.5*u1 : c   (later h = hb + q*ringmean, q = unc?+0.5:-1)
__global__ __launch_bounds__(256) void k_stage1(
    const float* __restrict__ img_re, const float* __restrict__ img_im,
    const int* __restrict__ mask, const int* __restrict__ ring_id,
    const float* __restrict__ w1_re, const float* __restrict__ w1_im,
    const float* __restrict__ w2_re, const float* __restrict__ w2_im,
    float* __restrict__ hb_re, float* __restrict__ hb_im,
    float* __restrict__ s_re, float* __restrict__ s_im) {
  int wave = threadIdx.x >> 5, lane = threadIdx.x & 31;
  int p0 = (blockIdx.x * 8 + wave) * 16;
  int hs = lane >> 4, col0 = lane & 15;

  __builtin_prefetch(w1_re, 0, 3);
  __builtin_prefetch(w1_im, 0, 3);
  __builtin_prefetch(w2_re, 0, 3);
  __builtin_prefetch(w2_im, 0, 3);

  // Load x once; A rows are per-lane uniform (row = lane&15 for both halves),
  // so the unc/cor masks are whole-fragment lane-scalar selects.
  v16h xr_h, xr_l, xi_h, xi_l;
  frag_split<false>([&](int m, int k) { return img_re[(p0 + m) * CIN + k]; },
                    lane, xr_h, xr_l);
  frag_split<false>([&](int m, int k) { return img_im[(p0 + m) * CIN + k]; },
                    lane, xi_h, xi_l);
  bool unc = (mask[p0 + (lane & 15)] == 0);
  v16h zh = {};
  v16h arU_h = unc ? xr_h : zh, arU_l = unc ? xr_l : zh;
  v16h aiU_h = unc ? xi_h : zh, aiU_l = unc ? xi_l : zh;
  v16h arC_h = unc ? zh : xr_h, arC_l = unc ? zh : xr_l;
  v16h aiC_h = unc ? zh : xi_h, aiC_l = unc ? zh : xi_l;

  for (int cb = 0; cb < 2; ++cb) {
    v16h b1r_h, b1r_l, b1i_h, b1i_l, b2r_h, b2r_l, b2i_h, b2i_l;
    frag_split<true>([&](int nn, int k) { return w1_re[k * COUT + cb * 16 + nn]; },
                     lane, b1r_h, b1r_l);
    frag_split<true>([&](int nn, int k) { return w1_im[k * COUT + cb * 16 + nn]; },
                     lane, b1i_h, b1i_l);
    frag_split<true>([&](int nn, int k) { return w2_re[k * COUT + cb * 16 + nn]; },
                     lane, b2r_h, b2r_l);
    frag_split<true>([&](int nn, int k) { return w2_im[k * COUT + cb * 16 + nn]; },
                     lane, b2i_h, b2i_l);

    v8f z = {};
    v8f u1re = wmma3(arU_h, arU_l, b1r_h, b1r_l, z);
    v8f t    = wmma3(aiU_h, aiU_l, b1i_h, b1i_l, z);
    u1re = u1re - t;
    v8f u1im = wmma3(arU_h, arU_l, b1i_h, b1i_l, z);
    u1im = wmma3(aiU_h, aiU_l, b1r_h, b1r_l, u1im);
    v8f cre = wmma3(arC_h, arC_l, b2r_h, b2r_l, z);
    t = wmma3(aiC_h, aiC_l, b2i_h, b2i_l, z);
    cre = cre - t;
    v8f cim = wmma3(arC_h, arC_l, b2i_h, b2i_l, z);
    cim = wmma3(aiC_h, aiC_l, b2r_h, b2r_l, cim);

#pragma unroll
    for (int v = 0; v < 8; ++v) {
      int row = v + (hs << 3);
      int p = p0 + row;
      int c = cb * 16 + col0;
      int r = ring_id[p];
      atomicAdd(&s_re[r * COUT + c], u1re[v]);
      atomicAdd(&s_im[r * COUT + c], u1im[v]);
      bool u = (mask[p] == 0);
      hb_re[p * COUT + c] = u ? 0.5f * u1re[v] : cre[v];
      hb_im[p * COUT + c] = u ? 0.5f * u1im[v] : cim[v];
    }
  }
}

// ---------------- stage 2: h, hc@L1 / hc@L2, per-ring Gram accumulation ---
// one wave per 32-pixel tile (two 16-row sub-tiles for the GEMM;
// 32 pixels = one K=32 WMMA contraction for the Gram).
__global__ __launch_bounds__(64) void k_att(
    const float* __restrict__ ro_theta, const int* __restrict__ mask,
    const int* __restrict__ ring_id,
    const float* __restrict__ L1_re, const float* __restrict__ L1_im,
    const float* __restrict__ L2_re, const float* __restrict__ L2_im,
    float* __restrict__ h_re, float* __restrict__ h_im,  // in: hb, out: h
    const float* __restrict__ s_re, const float* __restrict__ s_im,
    float* __restrict__ G_re, float* __restrict__ G_im) {
  __shared__ float ls[2][4][32][32];  // [wave][m1re,m1im,m2re,m2im][pix][feat]
  __shared__ int lrg[2][32];
  int wave = threadIdx.x >> 5, lane = threadIdx.x & 31;
  int p0 = (blockIdx.x * 2 + wave) * 32;
  int hs = lane >> 4, col0 = lane & 15;
  lrg[wave][lane] = ring_id[p0 + lane];

  v16h hre_h[2], hre_l[2], him_h[2], him_l[2];
#pragma unroll
  for (int rt = 0; rt < 2; ++rt) {
    frag_split<false>([&](int m, int k) {
        int p = p0 + rt * 16 + m;
        float q = (mask[p] == 0) ? 0.5f : -1.0f;
        float v = h_re[p * COUT + k] + q * s_re[ring_id[p] * COUT + k];
        h_re[p * COUT + k] = v;  // finalize h for the apply stage
        return v;
      }, lane, hre_h[rt], hre_l[rt]);
    frag_split<false>([&](int m, int k) {
        int p = p0 + rt * 16 + m;
        float q = (mask[p] == 0) ? 0.5f : -1.0f;
        float v = h_im[p * COUT + k] + q * s_im[ring_id[p] * COUT + k];
        h_im[p * COUT + k] = v;
        return v;
      }, lane, him_h[rt], him_l[rt]);
  }

  for (int cb = 0; cb < 2; ++cb) {
    v16h l1r_h, l1r_l, l1i_h, l1i_l, l2r_h, l2r_l, l2i_h, l2i_l;
    frag_split<true>([&](int nn, int k) { return L1_re[k * COUT + cb * 16 + nn]; },
                     lane, l1r_h, l1r_l);
    frag_split<true>([&](int nn, int k) { return L1_im[k * COUT + cb * 16 + nn]; },
                     lane, l1i_h, l1i_l);
    frag_split<true>([&](int nn, int k) { return L2_re[k * COUT + cb * 16 + nn]; },
                     lane, l2r_h, l2r_l);
    frag_split<true>([&](int nn, int k) { return L2_im[k * COUT + cb * 16 + nn]; },
                     lane, l2i_h, l2i_l);

#pragma unroll
    for (int rt = 0; rt < 2; ++rt) {
      v8f z = {};
      v8f m1re = wmma3(hre_h[rt], hre_l[rt], l1r_h, l1r_l, z);
      v8f t    = wmma3(him_h[rt], him_l[rt], l1i_h, l1i_l, z);
      m1re = m1re - t;
      v8f m1im = wmma3(hre_h[rt], hre_l[rt], l1i_h, l1i_l, z);
      m1im = wmma3(him_h[rt], him_l[rt], l1r_h, l1r_l, m1im);
      v8f m2re = wmma3(hre_h[rt], hre_l[rt], l2r_h, l2r_l, z);
      t = wmma3(him_h[rt], him_l[rt], l2i_h, l2i_l, z);
      m2re = m2re - t;
      v8f m2im = wmma3(hre_h[rt], hre_l[rt], l2i_h, l2i_l, z);
      m2im = wmma3(him_h[rt], him_l[rt], l2r_h, l2r_l, m2im);

#pragma unroll
      for (int v = 0; v < 8; ++v) {
        int row = v + (hs << 3);
        int p = p0 + rt * 16 + row;
        int c = cb * 16 + col0;
        float rr = ro_theta[p * 2 + 0], ph = ro_theta[p * 2 + 1];
        // rows 32,33 of L* (ro_theta channels, imaginary part zero)
        m1re[v] += rr * L1_re[32 * COUT + c] + ph * L1_re[33 * COUT + c];
        m1im[v] += rr * L1_im[32 * COUT + c] + ph * L1_im[33 * COUT + c];
        m2re[v] += rr * L2_re[32 * COUT + c] + ph * L2_re[33 * COUT + c];
        m2im[v] += rr * L2_im[32 * COUT + c] + ph * L2_im[33 * COUT + c];
        int pr = rt * 16 + row;
        ls[wave][0][pr][c] = m1re[v];
        ls[wave][1][pr][c] = m1im[v];
        ls[wave][2][pr][c] = m2re[v];
        ls[wave][3][pr][c] = m2im[v];
      }
    }
  }
  __syncthreads();

  // ----- per-ring Gram: G[r] += m1[sel]^T @ m2, pixel contraction K=32 -----
  int rmin = 1 << 30, rmax = -1;
  for (int i = 0; i < 32; ++i) {
    int r = lrg[wave][i];
    rmin = min(rmin, r);
    rmax = max(rmax, r);
  }
  rmin = __builtin_amdgcn_readfirstlane(rmin);
  rmax = __builtin_amdgcn_readfirstlane(rmax);

  // ring id of each K slot of this lane's A fragments (A map: K = pixel)
  int rgk[16];
#pragma unroll
  for (int i = 0; i < 16; ++i) rgk[i] = lrg[wave][kmapA(i, hs)];

  // unmasked m1 (A layout) and m2 (B layout) fragments, converted once
  v16h m1reF[2], m1imF[2], b2re[2], b2im[2];
#pragma unroll
  for (int ib = 0; ib < 2; ++ib) {
    m1reF[ib] = frag16<false>(
        [&](int f, int k) { return ls[wave][0][k][ib * 16 + f]; }, lane);
    m1imF[ib] = frag16<false>(
        [&](int f, int k) { return ls[wave][1][k][ib * 16 + f]; }, lane);
    b2re[ib] = frag16<true>(
        [&](int nn, int k) { return ls[wave][2][k][ib * 16 + nn]; }, lane);
    b2im[ib] = frag16<true>(
        [&](int nn, int k) { return ls[wave][3][k][ib * 16 + nn]; }, lane);
  }

  for (int r = rmin; r <= rmax; ++r) {  // scalar loop: EXEC stays full
    int any = 0;
    for (int i = 0; i < 32; ++i) any |= (lrg[wave][i] == r) ? 1 : 0;
    any = __builtin_amdgcn_readfirstlane(any);
    if (!any) continue;
    v16h a1re[2], a1im[2];
#pragma unroll
    for (int ib = 0; ib < 2; ++ib) {
#pragma unroll
      for (int i = 0; i < 16; ++i) {
        bool on = (rgk[i] == r);
        a1re[ib][i] = on ? m1reF[ib][i] : (_Float16)0.0f;
        a1im[ib][i] = on ? m1imF[ib][i] : (_Float16)0.0f;
      }
    }
#pragma unroll
    for (int ib = 0; ib < 2; ++ib) {
#pragma unroll
      for (int jb = 0; jb < 2; ++jb) {
        v8f z = {};
        v8f gre = wmma16(a1re[ib], b2re[jb], z) - wmma16(a1im[ib], b2im[jb], z);
        v8f gim = wmma16(a1re[ib], b2im[jb], z) + wmma16(a1im[ib], b2re[jb], z);
#pragma unroll
        for (int v = 0; v < 8; ++v) {
          int row = v + (hs << 3);
          int idx = r * (COUT * COUT) + (ib * 16 + row) * COUT + jb * 16 + col0;
          atomicAdd(&G_re[idx], gre[v]);
          atomicAdd(&G_im[idx], gim[v]);
        }
      }
    }
  }
}

// ---------------- stage 3: out = h @ M[ring] (complex) --------------------
__global__ __launch_bounds__(256) void k_apply(
    const float* __restrict__ h_re, const float* __restrict__ h_im,
    const int* __restrict__ ring_id,
    const float* __restrict__ M_re, const float* __restrict__ M_im,
    float* __restrict__ out) {
  __shared__ int lrg[8][16];
  int wave = threadIdx.x >> 5, lane = threadIdx.x & 31;
  int p0 = (blockIdx.x * 8 + wave) * 16;
  int hs = lane >> 4, col0 = lane & 15;
  if (lane < 16) lrg[wave][lane] = ring_id[p0 + lane];
  __syncthreads();

  int rmin = 1 << 30, rmax = -1;
  for (int i = 0; i < 16; ++i) {
    int r = lrg[wave][i];
    rmin = min(rmin, r);
    rmax = max(rmax, r);
  }
  rmin = __builtin_amdgcn_readfirstlane(rmin);
  rmax = __builtin_amdgcn_readfirstlane(rmax);

  // load h fragments once; A rows are per-lane uniform -> per-ring mask is a
  // whole-fragment lane-scalar select
  v16h hRfull = frag16<false>(
      [&](int m, int k) { return h_re[(p0 + m) * COUT + k]; }, lane);
  v16h hIfull = frag16<false>(
      [&](int m, int k) { return h_im[(p0 + m) * COUT + k]; }, lane);
  int rm = lrg[wave][lane & 15];
  v16h zh = {};

  v8f z = {};
  v8f acc_rr[2], acc_ii[2], acc_ri[2], acc_ir[2];
#pragma unroll
  for (int cb = 0; cb < 2; ++cb) {
    acc_rr[cb] = z; acc_ii[cb] = z; acc_ri[cb] = z; acc_ir[cb] = z;
  }
  for (int r = rmin; r <= rmax; ++r) {  // scalar loop: EXEC stays full
    int any = 0;
    for (int i = 0; i < 16; ++i) any |= (lrg[wave][i] == r) ? 1 : 0;
    any = __builtin_amdgcn_readfirstlane(any);
    if (!any) continue;
    v16h aR = (rm == r) ? hRfull : zh;
    v16h aI = (rm == r) ? hIfull : zh;
#pragma unroll
    for (int cb = 0; cb < 2; ++cb) {
      v16h bR = frag16<true>([&](int nn, int k) {
          return M_re[r * (COUT * COUT) + k * COUT + cb * 16 + nn];
        }, lane);
      v16h bI = frag16<true>([&](int nn, int k) {
          return M_im[r * (COUT * COUT) + k * COUT + cb * 16 + nn];
        }, lane);
      acc_rr[cb] = wmma16(aR, bR, acc_rr[cb]);
      acc_ii[cb] = wmma16(aI, bI, acc_ii[cb]);
      acc_ri[cb] = wmma16(aR, bI, acc_ri[cb]);
      acc_ir[cb] = wmma16(aI, bR, acc_ir[cb]);
    }
  }
#pragma unroll
  for (int cb = 0; cb < 2; ++cb) {
    v8f ore = acc_rr[cb] - acc_ii[cb];
    v8f oim = acc_ri[cb] + acc_ir[cb];
#pragma unroll
    for (int v = 0; v < 8; ++v) {
      int p = p0 + v + (hs << 3);
      int c = cb * 16 + col0;
      out[(p * COUT + c) * 2 + 0] = ore[v];  // complex64 interleaved
      out[(p * COUT + c) * 2 + 1] = oim[v];
    }
  }
}

extern "C" void kernel_launch(void* const* d_in, const int* in_sizes, int n_in,
                              void* d_out, int out_size, void* d_ws, size_t ws_size,
                              hipStream_t stream) {
  (void)in_sizes; (void)n_in; (void)out_size; (void)ws_size;
  const float* img_re   = (const float*)d_in[0];
  const float* img_im   = (const float*)d_in[1];
  const float* ro_theta = (const float*)d_in[2];
  const float* w1_re    = (const float*)d_in[3];
  const float* w1_im    = (const float*)d_in[4];
  const float* w2_re    = (const float*)d_in[5];
  const float* w2_im    = (const float*)d_in[6];
  const float* L1_re    = (const float*)d_in[7];
  const float* L1_im    = (const float*)d_in[8];
  const float* L2_re    = (const float*)d_in[9];
  const float* L2_im    = (const float*)d_in[10];
  const int*   mask     = (const int*)d_in[11];
  const int*   ring_id  = (const int*)d_in[12];

  float* ws    = (float*)d_ws;
  float* hb_re = ws;                         // NPIX*32
  float* hb_im = hb_re + NPIX * COUT;        // NPIX*32
  float* s_re  = hb_im + NPIX * COUT;        // 23*32
  float* s_im  = s_re + NRING * COUT;
  float* cnt   = s_im + NRING * COUT;        // 23 (padded to 32)
  float* G_re  = cnt + 32;                   // 23*32*32
  float* G_im  = G_re + NRING * COUT * COUT; // 23*32*32

  int zcount = 2 * NRING * COUT + 32 + 2 * NRING * COUT * COUT;
  k_zero<<<64, 256, 0, stream>>>(s_re, zcount);
  k_count<<<128, 256, 0, stream>>>(ring_id, cnt);
  k_stage1<<<NPIX / (16 * 8), 256, 0, stream>>>(
      img_re, img_im, mask, ring_id, w1_re, w1_im, w2_re, w2_im,
      hb_re, hb_im, s_re, s_im);
  k_mean<<<3, 256, 0, stream>>>(s_re, s_im, cnt);
  k_att<<<NPIX / (32 * 2), 64, 0, stream>>>(
      ro_theta, mask, ring_id, L1_re, L1_im, L2_re, L2_im,
      hb_re, hb_im, s_re, s_im, G_re, G_im);
  k_sigmoid<<<46, 256, 0, stream>>>(G_re, 2 * NRING * COUT * COUT);
  k_apply<<<NPIX / (16 * 8), 256, 0, stream>>>(
      hb_re, hb_im, ring_id, G_re, G_im, (float*)d_out);
}
